// SelfAttention_PostLN_71949292143045
// MI455X (gfx1250) — matrix operand order repaired and checked by reading, not compiled, not run on hardware
//
#include <hip/hip_runtime.h>

// ---------------------------------------------------------------------------
// MI455X (gfx1250, wave32) fused multi-head attention forward.
// - All GEMMs: v_wmma_f32_16x16x32_f16 (f16 in, f32 accumulate).
// - Attention computes S^T = K*Q^T so the WMMA C-layout of S^T is directly the
//   A-fragment layout of P for the P*V WMMA (no LDS transpose), and the
//   softmax kv-reduction needs only one cross-half shfl_xor(16).
// - K/V tiles staged to LDS with double buffering; uses CDNA5
//   GLOBAL_LOAD_ASYNC_TO_LDS_B128 + s_wait_asynccnt when available so the
//   HBM->LDS DMA overlaps the WMMA stream.
// ---------------------------------------------------------------------------

typedef _Float16 half_t;
typedef __attribute__((ext_vector_type(16))) _Float16 v16h;
typedef __attribute__((ext_vector_type(8)))  _Float16 v8h;
typedef __attribute__((ext_vector_type(8)))  float    v8f;
typedef __attribute__((ext_vector_type(4)))  float    v4f;
typedef __attribute__((ext_vector_type(4)))  int      v4i;

static constexpr int Bdim = 2, Ldim = 2048, Cdim = 1024, DKdim = 64, Hdim = 16;
static constexpr float kLog2e = 1.44269504088896340736f;
static_assert(Ldim == 2048 && Cdim == 1024 && DKdim == 64, "pow2 index math");

#if defined(__AMDGCN__) && __has_builtin(__builtin_amdgcn_global_load_async_to_lds_b128) && __has_builtin(__builtin_amdgcn_s_wait_asynccnt)
#define USE_ASYNC_LDS 1
#else
#define USE_ASYNC_LDS 0
#endif

#if USE_ASYNC_LDS
typedef __attribute__((address_space(1))) v4i as1_v4i;  // global int4
typedef __attribute__((address_space(3))) v4i as3_v4i;  // LDS int4
static __device__ __forceinline__ void stage16(const half_t* g, half_t* l) {
  __builtin_amdgcn_global_load_async_to_lds_b128((as1_v4i*)g, (as3_v4i*)l,
                                                 /*offset=*/0, /*cpol=*/0);
}
static __device__ __forceinline__ void stage_wait() {
  __builtin_amdgcn_s_wait_asynccnt(0);
}
#else
static __device__ __forceinline__ void stage16(const half_t* g, half_t* l) {
  *(v8h*)l = *(const v8h*)g;  // sync fallback: b128 load + ds_store
}
static __device__ __forceinline__ void stage_wait() {}
#endif

static __device__ __forceinline__ v8f vzero8() {
  v8f r;
#pragma unroll
  for (int i = 0; i < 8; ++i) r[i] = 0.f;
  return r;
}

static __device__ __forceinline__ v16h pack16(v8h lo, v8h hi) {
  v16h r;
#pragma unroll
  for (int i = 0; i < 8; ++i) { r[i] = lo[i]; r[8 + i] = hi[i]; }
  return r;
}

static __device__ __forceinline__ v8f wmma16(v16h a, v16h b, v8f c) {
  // (neg_a, A, neg_b, B, c_mod, C, reuse_a, reuse_b)
  return __builtin_amdgcn_wmma_f32_16x16x32_f16(false, a, false, b, (short)0, c,
                                                false, false);
}

// ---------------------------------------------------------------------------
__global__ __launch_bounds__(256) void cvt_f32_to_f16(const float* __restrict__ src,
                                                      half_t* __restrict__ dst,
                                                      unsigned n8) {
  for (unsigned i = blockIdx.x * blockDim.x + threadIdx.x; i < n8;
       i += gridDim.x * blockDim.x) {
    v4f a = ((const v4f*)src)[2 * i];
    v4f b = ((const v4f*)src)[2 * i + 1];
    v8h o;
#pragma unroll
    for (int e = 0; e < 4; ++e) { o[e] = (half_t)a[e]; o[4 + e] = (half_t)b[e]; }
    ((v8h*)dst)[i] = o;
  }
}

// ---------------------------------------------------------------------------
// Y = A[M,K] * Bw[N,K]^T + bias; M=B*L=4096, N=K=C=1024 (all four GEMMs).
// 128x64 block tile, 8 waves of 32x32.
// MODE 0: f16 out, [B,H,L,DK]   (Q and K; scale folds 1/sqrt(DK) into Q)
// MODE 2: f16 out, [B,H,DK,L]   (V transposed for contiguous PV B-fragments)
// MODE 3: f32 out, row-major [M,N]  (final output projection)
template <int MODE>
__global__ __launch_bounds__(256) void gemm_wmma(const half_t* __restrict__ A,
                                                 const half_t* __restrict__ Bw,
                                                 const float* __restrict__ bias,
                                                 void* __restrict__ outp,
                                                 float scale) {
  constexpr unsigned N = Cdim, K = Cdim;
  const unsigned wave = threadIdx.x >> 5;
  const unsigned lane = threadIdx.x & 31;
  const unsigned ln = lane & 15, lh = lane >> 4;
  const unsigned m0 = blockIdx.y * 128 + (wave & 3) * 32;
  const unsigned n0 = blockIdx.x * 64 + (wave >> 2) * 32;

  v8f acc[2][2];
#pragma unroll
  for (int i = 0; i < 2; ++i)
#pragma unroll
    for (int j = 0; j < 2; ++j) acc[i][j] = vzero8();

  for (unsigned k0 = 0; k0 < K; k0 += 32) {
    v16h af[2], bf[2];
#pragma unroll
    for (int i = 0; i < 2; ++i) {
      // A-fragment: lane = row m, elems e -> k = 16*lh + e  (contiguous 32B)
      const half_t* p = A + (size_t)(m0 + i * 16 + ln) * K + k0;
      af[i] = pack16(*(const v8h*)(p + 16 * lh), *(const v8h*)(p + 16 * lh + 8));
    }
#pragma unroll
    for (int j = 0; j < 2; ++j) {
      // B-fragment: lane = col n, elems e -> k = 16*lh + e  (contiguous 32B)
      const half_t* p = Bw + (size_t)(n0 + j * 16 + ln) * K + k0 + 16 * lh;
      bf[j] = pack16(*(const v8h*)p, *(const v8h*)(p + 8));
    }
#pragma unroll
    for (int i = 0; i < 2; ++i)
#pragma unroll
      for (int j = 0; j < 2; ++j) acc[i][j] = wmma16(af[i], bf[j], acc[i][j]);
  }

#pragma unroll
  for (int i = 0; i < 2; ++i)
#pragma unroll
    for (int j = 0; j < 2; ++j) {
      const unsigned ng = n0 + j * 16 + ln;
      const float bs = bias[ng];
#pragma unroll
      for (int r = 0; r < 8; ++r) {
        const unsigned mg = m0 + i * 16 + r + 8 * lh;  // D row = r + 8*(lane/16)
        const float v = (acc[i][j][r] + bs) * scale;
        if (MODE == 3) {
          ((float*)outp)[(size_t)mg * N + ng] = v;
        } else {
          const unsigned b = mg >> 11, l = mg & (Ldim - 1);   // Ldim = 2048
          const unsigned h = ng >> 6, d = ng & (DKdim - 1);   // DKdim = 64
          const size_t idx =
              (MODE == 0) ? ((size_t)(b * Hdim + h) * Ldim + l) * DKdim + d
                          : ((size_t)(b * Hdim + h) * DKdim + d) * Ldim + l;
          ((half_t*)outp)[idx] = (half_t)v;
        }
      }
    }
}

// ---------------------------------------------------------------------------
// Flash attention: each wave owns 32 q-rows; workgroup (8 waves) = 256 q-rows.
// Double-buffered LDS tiles: K (32x64) and V^T (64x32) per kv step, prefetched
// asynchronously one step ahead.
__global__ __launch_bounds__(256) void attn_wmma(const half_t* __restrict__ Q,
                                                 const half_t* __restrict__ Kh,
                                                 const half_t* __restrict__ VT,
                                                 const int* __restrict__ mask,
                                                 half_t* __restrict__ Z) {
  __shared__ half_t sK[2][32][72];  // [buf][kv][d], 144B rows (16B aligned)
  __shared__ half_t sV[2][64][40];  // [buf][d][kv],  80B rows (16B aligned)

  const unsigned wave = threadIdx.x >> 5;
  const unsigned lane = threadIdx.x & 31;
  const unsigned ln = lane & 15, lh = lane >> 4;
  const unsigned bh = blockIdx.y;
  const unsigned b = bh >> 4, h = bh & (Hdim - 1);
  const unsigned q0 = blockIdx.x * 256 + wave * 32;

  const half_t* Qb = Q + (size_t)bh * Ldim * DKdim;
  const half_t* Kb = Kh + (size_t)bh * Ldim * DKdim;
  const half_t* Vb = VT + (size_t)bh * DKdim * Ldim;
  const int* mb = mask + (size_t)b * Ldim * Ldim;

  // per-thread 16B staging chunk coordinates
  const unsigned t_ = threadIdx.x;
  const unsigned kr = t_ >> 3, kc = (t_ & 7) << 3;  // K tile: 32 x 64 halfs
  const unsigned vr = t_ >> 2, vc = (t_ & 3) << 3;  // V^T tile: 64 x 32 halfs

  auto stage = [&](unsigned buf, unsigned j0) {
    stage16(Kb + (size_t)(j0 + kr) * DKdim + kc, &sK[buf][kr][kc]);
    stage16(Vb + (size_t)vr * Ldim + j0 + vc, &sV[buf][vr][vc]);
  };

  // Q fragments in B-layout (Q is the B operand of S^T = K * Q^T).
  v16h qf[2][2];
#pragma unroll
  for (int qi = 0; qi < 2; ++qi)
#pragma unroll
    for (int ks = 0; ks < 2; ++ks) {
      const half_t* p =
          Qb + (size_t)(q0 + qi * 16 + ln) * DKdim + ks * 32 + 16 * lh;
      qf[qi][ks] = pack16(*(const v8h*)p, *(const v8h*)(p + 8));
    }

  v8f o[2][4];
#pragma unroll
  for (int qi = 0; qi < 2; ++qi)
#pragma unroll
    for (int dj = 0; dj < 4; ++dj) o[qi][dj] = vzero8();
  float mrow[2] = {-1e30f, -1e30f};
  float lrow[2] = {0.f, 0.f};

  constexpr int nT = Ldim / 32;
  stage(0, 0);
  for (int t = 0; t < nT; ++t) {
    stage_wait();      // our async loads for tile t have landed in LDS
    __syncthreads();   // everyone's have; buf (t^1) is also free to overwrite
    const unsigned cur = t & 1;
    const unsigned j0 = (unsigned)t * 32;
    if (t + 1 < nT) stage(cur ^ 1, j0 + 32);  // prefetch under the WMMAs

    v16h kf[2][2], vf[4];
#pragma unroll
    for (int kvi = 0; kvi < 2; ++kvi)
#pragma unroll
      for (int ks = 0; ks < 2; ++ks)
        kf[kvi][ks] =
            pack16(*(const v8h*)&sK[cur][kvi * 16 + ln][ks * 32 + 16 * lh],
                   *(const v8h*)&sK[cur][kvi * 16 + ln][ks * 32 + 16 * lh + 8]);
#pragma unroll
    for (int dj = 0; dj < 4; ++dj)
      vf[dj] = pack16(*(const v8h*)&sV[cur][dj * 16 + ln][8 * lh],
                      *(const v8h*)&sV[cur][dj * 16 + ln][16 + 8 * lh]);

    // S^T = K * Q^T : rows = kv, cols = q
    v8f s[2][2];
#pragma unroll
    for (int kvi = 0; kvi < 2; ++kvi)
#pragma unroll
      for (int qi = 0; qi < 2; ++qi) {
        s[kvi][qi] = vzero8();
#pragma unroll
        for (int ks = 0; ks < 2; ++ks)
          s[kvi][qi] = wmma16(kf[kvi][ks], qf[qi][ks], s[kvi][qi]);
      }

    // mask: S^T elem r has kv = j0 + kvi*16 + 8*lh + r -> 8 contiguous ints
#pragma unroll
    for (int kvi = 0; kvi < 2; ++kvi)
#pragma unroll
      for (int qi = 0; qi < 2; ++qi) {
        const int* mp =
            mb + (size_t)(q0 + qi * 16 + ln) * Ldim + j0 + kvi * 16 + 8 * lh;
        v4i ma = *(const v4i*)mp;
        v4i mc = *(const v4i*)(mp + 4);
#pragma unroll
        for (int r = 0; r < 4; ++r) {
          if (ma[r] == 0) s[kvi][qi][r] = -32767.f;
          if (mc[r] == 0) s[kvi][qi][r + 4] = -32767.f;
        }
      }

    // online softmax per q column (stats in lane ln, replicated across halves)
#pragma unroll
    for (int qi = 0; qi < 2; ++qi) {
      float tm = -1e30f;
#pragma unroll
      for (int kvi = 0; kvi < 2; ++kvi)
#pragma unroll
        for (int r = 0; r < 8; ++r) tm = fmaxf(tm, s[kvi][qi][r]);
      tm = fmaxf(tm, __shfl_xor(tm, 16, 32));
      float mn = fmaxf(mrow[qi], tm);
      float alpha = __builtin_amdgcn_exp2f((mrow[qi] - mn) * kLog2e);
      mrow[qi] = mn;
      float ps = 0.f;
#pragma unroll
      for (int kvi = 0; kvi < 2; ++kvi)
#pragma unroll
        for (int r = 0; r < 8; ++r) {
          float p = __builtin_amdgcn_exp2f((s[kvi][qi][r] - mn) * kLog2e);
          s[kvi][qi][r] = p;
          ps += p;
        }
      ps += __shfl_xor(ps, 16, 32);
      lrow[qi] = lrow[qi] * alpha + ps;

      // rescale O: broadcast alpha from stat lane (q) to D-layout row position
      float ab[8];
#pragma unroll
      for (int r = 0; r < 8; ++r) ab[r] = __shfl(alpha, r + 8 * lh, 32);
#pragma unroll
      for (int dj = 0; dj < 4; ++dj)
#pragma unroll
        for (int r = 0; r < 8; ++r) o[qi][dj][r] *= ab[r];

      // P A-fragment is a pure register remap of the S^T C-layout
      v16h pf;
#pragma unroll
      for (int e = 0; e < 8; ++e) {
        pf[e] = (half_t)s[0][qi][e];      // kv = 8*lh + e
        pf[8 + e] = (half_t)s[1][qi][e];  // kv = 16 + 8*lh + e
      }
#pragma unroll
      for (int dj = 0; dj < 4; ++dj) o[qi][dj] = wmma16(pf, vf[dj], o[qi][dj]);
    }
  }

  // normalize and store Z (f16, row-major [B*L, C] for the output GEMM)
#pragma unroll
  for (int qi = 0; qi < 2; ++qi) {
    float linv = 1.f / lrow[qi];
    float lb[8];
#pragma unroll
    for (int r = 0; r < 8; ++r) lb[r] = __shfl(linv, r + 8 * lh, 32);
#pragma unroll
    for (int dj = 0; dj < 4; ++dj)
#pragma unroll
      for (int r = 0; r < 8; ++r) {
        const unsigned qg = q0 + qi * 16 + r + 8 * lh;
        const unsigned col = h * DKdim + dj * 16 + ln;
        Z[((size_t)b * Ldim + qg) * Cdim + col] = (half_t)(o[qi][dj][r] * lb[r]);
      }
  }
}

// ---------------------------------------------------------------------------
extern "C" void kernel_launch(void* const* d_in, const int* in_sizes, int n_in,
                              void* d_out, int out_size, void* d_ws,
                              size_t ws_size, hipStream_t stream) {
  const float* qx = (const float*)d_in[0];
  const float* kx = (const float*)d_in[1];
  const float* vx = (const float*)d_in[2];
  const int* maskPAD = (const int*)d_in[3];
  const float* WQ_w = (const float*)d_in[4];
  const float* WQ_b = (const float*)d_in[5];
  const float* WK_w = (const float*)d_in[6];
  const float* WK_b = (const float*)d_in[7];
  const float* WV_w = (const float*)d_in[8];
  const float* WV_b = (const float*)d_in[9];
  const float* WO_w = (const float*)d_in[10];
  const float* WO_b = (const float*)d_in[11];
  (void)in_sizes; (void)n_in; (void)out_size; (void)ws_size;

  const size_t nX = (size_t)Bdim * Ldim * Cdim;  // 4 Mi elems
  const size_t nW = (size_t)Cdim * Cdim;         // 1 Mi elems

  char* ws = (char*)d_ws;
  size_t off = 0;
  auto take = [&](size_t nelem) {
    half_t* p = (half_t*)(ws + off);
    off += ((nelem * sizeof(half_t)) + 255) & ~(size_t)255;
    return p;
  };
  half_t* hQX = take(nX);
  half_t* hKX = take(nX);
  half_t* hVX = take(nX);
  half_t* hWQ = take(nW);
  half_t* hWK = take(nW);
  half_t* hWV = take(nW);
  half_t* hWO = take(nW);
  half_t* Qd  = take(nX);  // [B,H,L,DK], pre-scaled by 1/sqrt(DK)
  half_t* Kd  = take(nX);  // [B,H,L,DK]
  half_t* VTd = take(nX);  // [B,H,DK,L]
  half_t* Zd  = take(nX);  // [B,L,C]

  cvt_f32_to_f16<<<512, 256, 0, stream>>>(qx, hQX, (unsigned)(nX >> 3));
  cvt_f32_to_f16<<<512, 256, 0, stream>>>(kx, hKX, (unsigned)(nX >> 3));
  cvt_f32_to_f16<<<512, 256, 0, stream>>>(vx, hVX, (unsigned)(nX >> 3));
  cvt_f32_to_f16<<<256, 256, 0, stream>>>(WQ_w, hWQ, (unsigned)(nW >> 3));
  cvt_f32_to_f16<<<256, 256, 0, stream>>>(WK_w, hWK, (unsigned)(nW >> 3));
  cvt_f32_to_f16<<<256, 256, 0, stream>>>(WV_w, hWV, (unsigned)(nW >> 3));
  cvt_f32_to_f16<<<256, 256, 0, stream>>>(WO_w, hWO, (unsigned)(nW >> 3));

  dim3 gg(Cdim / 64, (Bdim * Ldim) / 128);  // (16, 32)
  gemm_wmma<0><<<gg, 256, 0, stream>>>(hQX, hWQ, WQ_b, Qd, 0.125f);  // 1/sqrt(64)
  gemm_wmma<0><<<gg, 256, 0, stream>>>(hKX, hWK, WK_b, Kd, 1.0f);
  gemm_wmma<2><<<gg, 256, 0, stream>>>(hVX, hWV, WV_b, VTd, 1.0f);

  dim3 ga(Ldim / 256, Bdim * Hdim);  // (8, 32)
  attn_wmma<<<ga, 256, 0, stream>>>(Qd, Kd, VTd, maskPAD, Zd);

  gemm_wmma<3><<<gg, 256, 0, stream>>>(Zd, hWO, WO_b, (float*)d_out, 1.0f);
}
